// ReconstructModel_65893388256033
// MI455X (gfx1250) — compile-verified, matrix operand order checked by reading, not run on hardware
//
#include <hip/hip_runtime.h>

// GRU decoder, persistent per-batch-tile, bf16 WMMA with f32 accumulate.
// BATCH=2048, HID=512, IN=OUT=128, SEQ=128.
// BT=32: each wave computes two 16-row M-subtiles so every weight (B) fragment
// fetched from L2 is reused by two WMMAs. f32 h state lives in LDS to keep
// per-wave VGPR usage under 256 (8 waves must co-reside on one WGP).

#define BATCH 2048
#define HID 512
#define IN_DIM 128
#define OUT_DIM 128
#define SEQ 128
#define G3 (3 * HID)          // 1536
#define BT 32                 // batch rows per block (2 M-subtiles of 16)
#define NBLK (BATCH / BT)     // 64 blocks
#define HLD (HID + 8)         // padded LDS row (bf16 elems) for h
#define XLD (IN_DIM + 8)      // padded LDS row for x
#define HP  (HID + 4)         // padded LDS row (f32 elems) for h32

typedef __attribute__((ext_vector_type(16))) __bf16 bf16x16;
typedef __attribute__((ext_vector_type(8)))  __bf16 bf16x8;
typedef __attribute__((ext_vector_type(8)))  float  floatx8;

__device__ __forceinline__ unsigned short f2bf(float f) {
  // round-to-nearest-even f32 -> bf16 bits
  unsigned u = __float_as_uint(f);
  u += 0x7FFFu + ((u >> 16) & 1u);
  return (unsigned short)(u >> 16);
}

// One lane's slice of a 16x32 bf16 WMMA operand: elements [0,8) at p,
// [8,16) at p+16 (CDNA5 16-bit layout: hi half-wave holds K+8 / K+24).
__device__ __forceinline__ bf16x16 load_frag(const unsigned short* p) {
  bf16x8 lo = *reinterpret_cast<const bf16x8*>(p);
  bf16x8 h8 = *reinterpret_cast<const bf16x8*>(p + 16);
  return __builtin_shufflevector(lo, h8, 0,1,2,3,4,5,6,7,8,9,10,11,12,13,14,15);
}

__device__ __forceinline__ floatx8 wmma_bf16(bf16x16 a, bf16x16 b, floatx8 c) {
  return __builtin_amdgcn_wmma_f32_16x16x32_bf16(false, a, false, b, (short)0, c,
                                                 false, false);
}

__device__ __forceinline__ float sigm(float x) {
  return 1.0f / (1.0f + __expf(-x));
}
__device__ __forceinline__ float tanh_(float x) {
  x = fminf(fmaxf(x, -15.0f), 15.0f);
  float e = __expf(-2.0f * x);
  return (1.0f - e) / (1.0f + e);
}

__device__ __forceinline__ void opt_fence() {
  // Prevent LICM from hoisting (and spilling) loop-invariant weight loads.
  asm volatile("" ::: "memory");
}
__device__ __forceinline__ void sched_fence() {
#if __has_builtin(__builtin_amdgcn_sched_barrier)
  __builtin_amdgcn_sched_barrier(0);  // cap cross-tile interleaving / pressure
#endif
}

__global__ void f32_to_bf16_kernel(const float* __restrict__ src,
                                   unsigned short* __restrict__ dst, int n) {
  int i = blockIdx.x * blockDim.x + threadIdx.x;
  if (i < n) dst[i] = f2bf(src[i]);
}

__global__ __launch_bounds__(256)
void gru_decoder_kernel(const float* __restrict__ hidden,       // (1,2048,512) f32
                        const float* __restrict__ b_ih,         // (1536,) f32
                        const float* __restrict__ b_hh,         // (1536,) f32
                        const float* __restrict__ fc_b,         // (128,) f32
                        const unsigned short* __restrict__ wih, // (1536,128) bf16
                        const unsigned short* __restrict__ whh, // (1536,512) bf16
                        const unsigned short* __restrict__ fcw, // (128,512) bf16
                        float* __restrict__ out)                // (2048,128,128) f32
{
  __shared__ __align__(16) unsigned short hls[2][BT][HLD]; // bf16 h, double-buffered
  __shared__ __align__(16) unsigned short xls[BT][XLD];    // bf16 x (feedback input)
  __shared__ __align__(16) float          h32[BT][HP];     // f32 h state

  const int tid   = threadIdx.x;
  const int wave  = tid >> 5;
  const int lane  = tid & 31;
  const int n     = lane & 15;   // A row / B column / C column index
  const int hi    = lane >> 4;   // 0: K {0..7,16..23}, 1: K {8..15,24..31}
  const int hioff = hi * 8;
  const int b0    = blockIdx.x * BT;

  // ---- per-lane base pointers; all per-fragment offsets are compile-time ----
  const unsigned short* wih_b = wih + ((size_t)wave * 64 + n) * IN_DIM + hioff;
  const unsigned short* whh_b = whh + ((size_t)wave * 64 + n) * HID   + hioff;
  const unsigned short* fcw_b = fcw + ((size_t)(wave * 16 + n)) * HID + hioff;
  unsigned short* hls0 = &hls[0][0][0];
  const int aoff  = n * HLD + hioff;               // A-frag base (h), subtile0
  const int stoff = 8 * hi * HLD + wave * 64 + n;  // h_new bf16 store base
  float* h32b = &h32[0][0] + 8 * hi * HP + wave * 64 + n;   // f32 h state base
  unsigned short* xstb = &xls[0][0] + 8 * hi * XLD + wave * 16 + n;
  const unsigned short* xA = &xls[0][0] + n * XLD + hioff;
  // output pointer, walks backwards one seq row per step (out[:, ::-1, :])
  float* outp = out + ((size_t)(b0 + 8 * hi) * SEQ + (SEQ - 1)) * OUT_DIM
                    + wave * 16 + n;

  // x0 = 0
  for (int i = tid; i < BT * XLD; i += 256) ((unsigned short*)xls)[i] = 0;

  // h state in f32 LDS; wave owns hidden cols [(wave*4+tt)*16,+16) for both
  // M-subtiles. Biases folded per lane.
  float br[4], bz[4], bin[4], bhn[4];
#pragma unroll
  for (int tt = 0; tt < 4; ++tt) {
    const int c0 = wave * 64 + tt * 16;
    br[tt]  = b_ih[c0 + n] + b_hh[c0 + n];
    bz[tt]  = b_ih[HID + c0 + n] + b_hh[HID + c0 + n];
    bin[tt] = b_ih[2 * HID + c0 + n];
    bhn[tt] = b_hh[2 * HID + c0 + n];
#pragma unroll
    for (int m = 0; m < 2; ++m) {
#pragma unroll
      for (int v = 0; v < 8; ++v) {
        const int M = m * 16 + 8 * hi + v;           // row within batch tile
        float h = hidden[(size_t)(b0 + M) * HID + c0 + n];
        h32b[(m * 16 + v) * HP + tt * 16] = h;
        hls0[stoff + (m * 16 + v) * HLD + tt * 16] = f2bf(h);
      }
    }
  }
  const float fb = fc_b[wave * 16 + n];
  __syncthreads();

  for (int t = 0; t < SEQ; ++t) {
    opt_fence();   // keep weight loads inside the loop (L2-resident streaming)
    const int cur = t & 1, nxt = cur ^ 1;
    const unsigned short* hA  = hls0 + cur * (BT * HLD) + aoff;
    const unsigned short* hAn = hls0 + nxt * (BT * HLD) + aoff;
    unsigned short*       hst = hls0 + nxt * (BT * HLD) + stoff;

    // ---------------- gate phase: r, z, n-gate, h update ----------------
#pragma unroll
    for (int tt = 0; tt < 4; ++tt) {
      sched_fence();
      floatx8 ar[2], az[2], ain[2], ahn[2];
#pragma unroll
      for (int m = 0; m < 2; ++m) {
        ar[m] = (floatx8)0.0f; az[m] = (floatx8)0.0f;
        ain[m] = (floatx8)0.0f; ahn[m] = (floatx8)0.0f;
      }
      // x contribution, K = 128; each B fragment feeds both M-subtiles
#pragma unroll 2
      for (int kf = 0; kf < IN_DIM / 32; ++kf) {
        const int k0 = kf * 32;
        bf16x16 a0 = load_frag(xA + k0);
        bf16x16 a1 = load_frag(xA + 16 * XLD + k0);
        bf16x16 brf = load_frag(wih_b + (          tt * 16) * IN_DIM + k0);
        ar[0] = wmma_bf16(a0, brf, ar[0]);  ar[1] = wmma_bf16(a1, brf, ar[1]);
        bf16x16 bzf = load_frag(wih_b + (    HID + tt * 16) * IN_DIM + k0);
        az[0] = wmma_bf16(a0, bzf, az[0]);  az[1] = wmma_bf16(a1, bzf, az[1]);
        bf16x16 bnf = load_frag(wih_b + (2 * HID + tt * 16) * IN_DIM + k0);
        ain[0] = wmma_bf16(a0, bnf, ain[0]); ain[1] = wmma_bf16(a1, bnf, ain[1]);
      }
      // h contribution, K = 512
#pragma unroll 2
      for (int kf = 0; kf < HID / 32; ++kf) {
        const int k0 = kf * 32;
        bf16x16 a0 = load_frag(hA + k0);
        bf16x16 a1 = load_frag(hA + 16 * HLD + k0);
        bf16x16 brf = load_frag(whh_b + (          tt * 16) * HID + k0);
        ar[0] = wmma_bf16(a0, brf, ar[0]);  ar[1] = wmma_bf16(a1, brf, ar[1]);
        bf16x16 bzf = load_frag(whh_b + (    HID + tt * 16) * HID + k0);
        az[0] = wmma_bf16(a0, bzf, az[0]);  az[1] = wmma_bf16(a1, bzf, az[1]);
        bf16x16 bnf = load_frag(whh_b + (2 * HID + tt * 16) * HID + k0);
        ahn[0] = wmma_bf16(a0, bnf, ahn[0]); ahn[1] = wmma_bf16(a1, bnf, ahn[1]);
      }
      // elementwise gates; f32 h state read/written in LDS (same lane only)
#pragma unroll
      for (int m = 0; m < 2; ++m) {
#pragma unroll
        for (int v = 0; v < 8; ++v) {
          const int ho = (m * 16 + v) * HP + tt * 16;
          float hold = h32b[ho];
          float r = sigm(ar[m][v] + br[tt]);
          float z = sigm(az[m][v] + bz[tt]);
          float g = tanh_(ain[m][v] + bin[tt] + r * (ahn[m][v] + bhn[tt]));
          float h = (1.0f - z) * g + z * hold;
          h32b[ho] = h;
          hst[(m * 16 + v) * HLD + tt * 16] = f2bf(h);
        }
      }
    }
    __syncthreads();   // all columns of h_new visible for fc GEMM

    // ---------------- fc phase: y = h_new·fc_wᵀ + fc_b ----------------
    {
      sched_fence();
      floatx8 ay0 = (floatx8)0.0f, ay1 = (floatx8)0.0f;
#pragma unroll 2
      for (int kf = 0; kf < HID / 32; ++kf) {
        const int k0 = kf * 32;
        bf16x16 a0 = load_frag(hAn + k0);
        bf16x16 a1 = load_frag(hAn + 16 * HLD + k0);
        bf16x16 bf = load_frag(fcw_b + k0);
        ay0 = wmma_bf16(a0, bf, ay0);
        ay1 = wmma_bf16(a1, bf, ay1);
      }
#pragma unroll
      for (int v = 0; v < 8; ++v) {
        float y0 = ay0[v] + fb;
        float y1 = ay1[v] + fb;
        xstb[v * XLD] = f2bf(y0);                    // next step's x, rows 0..15
        xstb[(16 + v) * XLD] = f2bf(y1);             // rows 16..31
        outp[v * (SEQ * OUT_DIM)] = y0;
        outp[(16 + v) * (SEQ * OUT_DIM)] = y1;
      }
      outp -= OUT_DIM;
    }
    __syncthreads();   // x fully written before next gate phase reads it
  }
}

extern "C" void kernel_launch(void* const* d_in, const int* in_sizes, int n_in,
                              void* d_out, int out_size, void* d_ws, size_t ws_size,
                              hipStream_t stream) {
  const float* hidden = (const float*)d_in[0];
  const float* w_ih   = (const float*)d_in[1];
  const float* w_hh   = (const float*)d_in[2];
  const float* b_ih   = (const float*)d_in[3];
  const float* b_hh   = (const float*)d_in[4];
  const float* fc_w   = (const float*)d_in[5];
  const float* fc_b   = (const float*)d_in[6];
  float* out = (float*)d_out;

  // bf16 weight shadows in workspace (2 MB total)
  unsigned short* wih_bf = (unsigned short*)d_ws;
  unsigned short* whh_bf = wih_bf + (size_t)G3 * IN_DIM;   // +196608 elems
  unsigned short* fcw_bf = whh_bf + (size_t)G3 * HID;      // +786432 elems

  const int CN1 = G3 * IN_DIM;     // 196608
  const int CN2 = G3 * HID;        // 786432
  const int CN3 = OUT_DIM * HID;   // 65536
  f32_to_bf16_kernel<<<(CN1 + 255) / 256, 256, 0, stream>>>(w_ih, wih_bf, CN1);
  f32_to_bf16_kernel<<<(CN2 + 255) / 256, 256, 0, stream>>>(w_hh, whh_bf, CN2);
  f32_to_bf16_kernel<<<(CN3 + 255) / 256, 256, 0, stream>>>(fc_w, fcw_bf, CN3);

  gru_decoder_kernel<<<NBLK, 256, 0, stream>>>(hidden, b_ih, b_hh, fc_b,
                                               wih_bf, whh_bf, fcw_bf, out);
}
